// ModelBaseLine_6167573037621
// MI455X (gfx1250) — matrix operand order repaired and checked by reading, not compiled, not run on hardware
//
#include <hip/hip_runtime.h>
#include <hip/hip_bf16.h>

// ---------------------------------------------------------------------------
// BERT-like encoder forward for MI455X (gfx1250), bf16 WMMA everywhere.
// B=8 S=512 D=768 H=12 L=6 I=3072 DH=64
// GEMMs take B in transposed [N,K] layout so every WMMA fragment is built
// from contiguous 128-bit LDS loads. Global->LDS staging uses the CDNA5
// async-to-LDS DMA path (ASYNCcnt) when the builtin is available.
// ---------------------------------------------------------------------------

typedef __attribute__((ext_vector_type(16))) __bf16 v16bf;
typedef __attribute__((ext_vector_type(8)))  float  v8f;

// builtin-compatible 16-byte int vector (HIP's int4 is a struct, not a vector)
typedef __attribute__((__vector_size__(16))) int vi4;
typedef __attribute__((address_space(3))) vi4 vi4_lds;

#define Bx 8
#define Sx 512
#define Dx 768
#define Hx 12
#define Lx 6
#define Ix 3072
#define DHx 64

#if __has_builtin(__builtin_amdgcn_global_load_async_to_lds_b128)
#define USE_ASYNC_LDS 1
#else
#define USE_ASYNC_LDS 0
#endif

// global source: plain generic vector pointer (matches builtin's param type)
#define GLB_V4(p) ((vi4*)(void*)(p))
// LDS dest: addrspace(3) pointer built from the low 32 bits of the generic
// shared address (== LDS byte offset per CDNA5 ISA 10.2)
#define LDS_V4(p) ((vi4_lds*)(unsigned)(unsigned long long)(void*)(p))

__device__ __forceinline__ void wait_async_lds()
{
#if USE_ASYNC_LDS
#if __has_builtin(__builtin_amdgcn_s_wait_asynccnt)
  __builtin_amdgcn_s_wait_asynccnt(0);
#else
  asm volatile("s_wait_asynccnt 0x0" ::: "memory");
#endif
#endif
}

// ---------------------------------------------------------------------------
// Batched bf16 GEMM, B transposed:  C[z] = A[z](MxK,row) * BT[z](NxK,row)^T
// BM=128 fixed. 256 threads = 8 waves (WR x WC), each wave computes TM x TN
// 16x16 WMMA tiles. BK=32. Double-buffered LDS; staging via async-to-LDS DMA
// (no VGPR residency) or short-lived load/store pairs. Padded LDS stride (40)
// => conflict-free b128 fragment reads.
// ---------------------------------------------------------------------------
template <int BN, int WR, int WC>
__global__ __launch_bounds__(256) void gemm_bt_kernel(
    const __bf16* __restrict__ Ag, const __bf16* __restrict__ BTg,
    const float* __restrict__ bias,
    float* __restrict__ Cf, __bf16* __restrict__ Cb,
    int M, int N, int K,
    long long sA, long long sB, long long sC, int relu)
{
  constexpr int TM  = 128 / (WR * 16);
  constexpr int TN  = BN  / (WC * 16);
  constexpr int LDA = 40;                 // padded K-stride in elements
  constexpr int NB  = BN / 64;            // B rows per thread (b128 chunks)

  __shared__ alignas(16) __bf16 As[2][128 * LDA];
  __shared__ alignas(16) __bf16 Bs[2][BN * LDA];

  const int z = blockIdx.z;
  const __bf16* A  = Ag  + (long long)z * sA;
  const __bf16* BT = BTg + (long long)z * sB;

  const int tid  = threadIdx.x;
  const int lane = tid & 31;
  const int wave = tid >> 5;
  const int wr   = wave / WC;
  const int wc   = wave % WC;
  const int l15  = lane & 15;
  const int half = lane >> 4;

  const int bm = blockIdx.y * 128;
  const int bn = blockIdx.x * BN;

  const int lrow = tid >> 2;              // 0..63
  const int lc4  = (tid & 3) * 8;         // 0,8,16,24 (elements)

  // stage one 128x32 A tile + BNx32 BT tile into LDS buffer `buf`
  auto stage = [&](int buf, int k0) {
#if USE_ASYNC_LDS
    __builtin_amdgcn_global_load_async_to_lds_b128(
        GLB_V4(&A[(long long)(bm + lrow) * K + k0 + lc4]),
        LDS_V4(&As[buf][lrow * LDA + lc4]), 0, 0);
    __builtin_amdgcn_global_load_async_to_lds_b128(
        GLB_V4(&A[(long long)(bm + lrow + 64) * K + k0 + lc4]),
        LDS_V4(&As[buf][(lrow + 64) * LDA + lc4]), 0, 0);
#pragma unroll
    for (int j = 0; j < NB; ++j)
      __builtin_amdgcn_global_load_async_to_lds_b128(
          GLB_V4(&BT[(long long)(bn + lrow + 64 * j) * K + k0 + lc4]),
          LDS_V4(&Bs[buf][(lrow + 64 * j) * LDA + lc4]), 0, 0);
#else
    uint4 r0 = *reinterpret_cast<const uint4*>(&A[(long long)(bm + lrow) * K + k0 + lc4]);
    uint4 r1 = *reinterpret_cast<const uint4*>(&A[(long long)(bm + lrow + 64) * K + k0 + lc4]);
    *reinterpret_cast<uint4*>(&As[buf][lrow * LDA + lc4]) = r0;
    *reinterpret_cast<uint4*>(&As[buf][(lrow + 64) * LDA + lc4]) = r1;
#pragma unroll
    for (int j = 0; j < NB; ++j) {
      uint4 rb = *reinterpret_cast<const uint4*>(&BT[(long long)(bn + lrow + 64 * j) * K + k0 + lc4]);
      *reinterpret_cast<uint4*>(&Bs[buf][(lrow + 64 * j) * LDA + lc4]) = rb;
    }
#endif
  };

  v8f acc[TM][TN] = {};

  stage(0, 0);
  int cur = 0;
  const int nt = K / 32;

  for (int t = 0; t < nt; ++t) {
    wait_async_lds();        // our async writes into buf[cur] are done
    __syncthreads();         // everyone's writes visible, prior reads retired
    if (t + 1 < nt) stage(cur ^ 1, (t + 1) * 32);

    v16bf af[TM], bf[TN];
#pragma unroll
    for (int m = 0; m < TM; ++m) {
      const int ar = (wr * TM + m) * 16 + l15;
#pragma unroll
      for (int i = 0; i < 8; ++i) {
        af[m][i]     = As[cur][ar * LDA + half * 8 + i];        // K {0-7 | 8-15}
        af[m][i + 8] = As[cur][ar * LDA + 16 + half * 8 + i];   // K {16-23 | 24-31}
      }
    }
#pragma unroll
    for (int n = 0; n < TN; ++n) {
      const int br = (wc * TN + n) * 16 + l15;
#pragma unroll
      for (int i = 0; i < 8; ++i) {
        bf[n][i]     = Bs[cur][br * LDA + half * 8 + i];        // K {0-7 | 8-15}
        bf[n][i + 8] = Bs[cur][br * LDA + 16 + half * 8 + i];   // K {16-23 | 24-31}
      }
    }

#pragma unroll
    for (int m = 0; m < TM; ++m)
#pragma unroll
      for (int n = 0; n < TN; ++n)
        acc[m][n] = __builtin_amdgcn_wmma_f32_16x16x32_bf16(
            false, af[m], false, bf[n], (short)0, acc[m][n], false, false);

    cur ^= 1;
  }

  const long long cbase = (long long)z * sC;
#pragma unroll
  for (int m = 0; m < TM; ++m) {
#pragma unroll
    for (int n = 0; n < TN; ++n) {
      const int col = bn + (wc * TN + n) * 16 + l15;
      const float bv = bias ? bias[col] : 0.0f;
#pragma unroll
      for (int r = 0; r < 8; ++r) {
        const int row = bm + (wr * TM + m) * 16 + r + half * 8;
        float v = acc[m][n][r] + bv;
        if (relu) v = fmaxf(v, 0.0f);
        const long long idx = cbase + (long long)row * N + col;
        if (Cf) Cf[idx] = v;
        if (Cb) Cb[idx] = (__bf16)v;
      }
    }
  }
}

// ---------------------------------------------------------------------------
// fp32 [L,K,N] -> bf16 [L,N,K] tiled transpose (coalesced both sides)
// ---------------------------------------------------------------------------
__global__ __launch_bounds__(256) void transpose_f2b_kernel(
    const float* __restrict__ in, __bf16* __restrict__ out, int K, int N)
{
  __shared__ float t[32][33];
  const int l  = blockIdx.z;
  const int kb = blockIdx.y * 32;
  const int nb = blockIdx.x * 32;
  const float* src = in  + (long long)l * K * N;
  __bf16*      dst = out + (long long)l * K * N;
  const int tx = threadIdx.x & 31;
  const int ty = threadIdx.x >> 5;   // 0..7
#pragma unroll
  for (int j = 0; j < 4; ++j)
    t[ty + j * 8][tx] = src[(long long)(kb + ty + j * 8) * N + nb + tx];
  __syncthreads();
#pragma unroll
  for (int j = 0; j < 4; ++j)
    dst[(long long)(nb + ty + j * 8) * K + kb + tx] = (__bf16)t[tx][ty + j * 8];
}

// ---------------------------------------------------------------------------
// Embedding gather: word + segment + position
// ---------------------------------------------------------------------------
__global__ void embed_kernel(const int* __restrict__ ids, const int* __restrict__ seg,
                             const float* __restrict__ we, const float* __restrict__ se,
                             const float* __restrict__ pe, float* __restrict__ out)
{
  const long long i = (long long)blockIdx.x * blockDim.x + threadIdx.x;
  const long long n = (long long)Bx * Sx * Dx;
  if (i >= n) return;
  const int d = (int)(i % Dx);
  const long long tok = i / Dx;
  const int s = (int)(tok % Sx);
  const int id = ids[tok], sg = seg[tok];
  out[i] = we[(long long)id * Dx + d] + se[(long long)sg * Dx + d] + pe[s * Dx + d];
}

// ---------------------------------------------------------------------------
// 2-D layernorm over the (S,D) slab per batch element (matches ln2d):
// y = in1 (+ in2), out = (y - mean) * rsqrt(var + eps) * g + b
// ---------------------------------------------------------------------------
__global__ __launch_bounds__(512) void ln2d_kernel(
    const float* __restrict__ in1, const float* __restrict__ in2,
    const float* __restrict__ g, const float* __restrict__ bt,
    float* __restrict__ outf, __bf16* __restrict__ outb)
{
  __shared__ float s1[512];
  __shared__ float s2[512];
  const int b = blockIdx.x;
  const int n = Sx * Dx;
  const long long base = (long long)b * n;

  float sum = 0.0f, sq = 0.0f;
  for (int i = threadIdx.x; i < n; i += 512) {
    float y = in1[base + i] + (in2 ? in2[base + i] : 0.0f);
    sum += y; sq += y * y;
  }
  s1[threadIdx.x] = sum; s2[threadIdx.x] = sq;
  __syncthreads();
  for (int st = 256; st > 0; st >>= 1) {
    if (threadIdx.x < st) {
      s1[threadIdx.x] += s1[threadIdx.x + st];
      s2[threadIdx.x] += s2[threadIdx.x + st];
    }
    __syncthreads();
  }
  const float mu  = s1[0] / (float)n;
  const float var = s2[0] / (float)n - mu * mu;
  const float rs  = rsqrtf(var + 1e-5f);

  for (int i = threadIdx.x; i < n; i += 512) {
    float y = in1[base + i] + (in2 ? in2[base + i] : 0.0f);
    float o = (y - mu) * rs * g[i] + bt[i];
    outf[base + i] = o;
    outb[base + i] = (__bf16)o;
  }
}

// ---------------------------------------------------------------------------
// Repack QKV [B,S,H*DH] bf16 -> q,k [B,H,S,DH]; v -> [B,H,DH,S] (transposed,
// because the context GEMM consumes B in [N,K] layout).
// ---------------------------------------------------------------------------
__global__ void qkv_repack_kernel(const __bf16* __restrict__ qt, const __bf16* __restrict__ kt,
                                  const __bf16* __restrict__ vt,
                                  __bf16* __restrict__ qh, __bf16* __restrict__ kh,
                                  __bf16* __restrict__ vT)
{
  const long long i = (long long)blockIdx.x * blockDim.x + threadIdx.x;
  const long long n = (long long)Bx * Sx * Dx;
  if (i >= n) return;
  const int d = (int)(i % Dx);
  const long long bs = i / Dx;
  const int s = (int)(bs % Sx);
  const int b = (int)(bs / Sx);
  const int h = d >> 6, dd = d & 63;
  const long long o = (((long long)(b * Hx + h)) * Sx + s) * DHx + dd;
  qh[o] = qt[i];
  kh[o] = kt[i];
  vT[(((long long)(b * Hx + h)) * DHx + dd) * Sx + s] = vt[i];
}

// ---------------------------------------------------------------------------
// Repack attention context [B,H,S,DH] -> [B,S,H*DH]
// ---------------------------------------------------------------------------
__global__ void attn_repack_kernel(const __bf16* __restrict__ ah, __bf16* __restrict__ ctx)
{
  const long long i = (long long)blockIdx.x * blockDim.x + threadIdx.x;
  const long long n = (long long)Bx * Sx * Dx;
  if (i >= n) return;
  const int d = (int)(i % Dx);
  const long long bs = i / Dx;
  const int s = (int)(bs % Sx);
  const int b = (int)(bs / Sx);
  const int h = d >> 6, dd = d & 63;
  ctx[i] = ah[(((long long)(b * Hx + h)) * Sx + s) * DHx + dd];
}

// ---------------------------------------------------------------------------
// Masked, scaled softmax over keys. One block per (b,h,q) row, bf16 output.
// ---------------------------------------------------------------------------
__global__ __launch_bounds__(256) void softmax_kernel(
    const float* __restrict__ scores, const float* __restrict__ mask,
    __bf16* __restrict__ probs, float scale)
{
  __shared__ float red[256];
  const int row = blockIdx.x;          // (b*H + h)*S + q
  const int q  = row & (Sx - 1);
  const int bh = row >> 9;
  const int b  = bh / Hx;
  const long long base = (long long)row * Sx;
  const float m = mask[b * Sx + q];

  float v0 = scores[base + threadIdx.x]       * scale * m;
  float v1 = scores[base + threadIdx.x + 256] * scale * m;

  red[threadIdx.x] = fmaxf(v0, v1);
  __syncthreads();
  for (int st = 128; st > 0; st >>= 1) {
    if (threadIdx.x < st) red[threadIdx.x] = fmaxf(red[threadIdx.x], red[threadIdx.x + st]);
    __syncthreads();
  }
  const float mx = red[0];
  __syncthreads();

  const float e0 = __expf(v0 - mx);
  const float e1 = __expf(v1 - mx);
  red[threadIdx.x] = e0 + e1;
  __syncthreads();
  for (int st = 128; st > 0; st >>= 1) {
    if (threadIdx.x < st) red[threadIdx.x] += red[threadIdx.x + st];
    __syncthreads();
  }
  const float inv = 1.0f / red[0];
  probs[base + threadIdx.x]       = (__bf16)(e0 * inv);
  probs[base + threadIdx.x + 256] = (__bf16)(e1 * inv);
}

// ---------------------------------------------------------------------------
// Final projection D -> 2 plus mask penalty. Output: [2, B*S] planes.
// ---------------------------------------------------------------------------
__global__ void final_kernel(const float* __restrict__ x, const float* __restrict__ Wp,
                             const float* __restrict__ bp, const float* __restrict__ mask,
                             float* __restrict__ out)
{
  const int idx = blockIdx.x * blockDim.x + threadIdx.x;
  if (idx >= 2 * Bx * Sx) return;
  const int t  = idx >> 1;
  const int ch = idx & 1;
  float acc = bp[ch];
  const float* xr = x + (long long)t * Dx;
  for (int d = 0; d < Dx; ++d) acc += xr[d] * Wp[d * 2 + ch];
  acc += (1.0f - mask[t]) * -10000.0f;
  out[ch * (Bx * Sx) + t] = acc;
}

// ---------------------------------------------------------------------------
// Host orchestration
// ---------------------------------------------------------------------------
extern "C" void kernel_launch(void* const* d_in, const int* in_sizes, int n_in,
                              void* d_out, int out_size, void* d_ws, size_t ws_size,
                              hipStream_t stream)
{
  (void)in_sizes; (void)n_in; (void)out_size; (void)ws_size;

  const int* ids    = (const int*)d_in[0];
  const int* seg    = (const int*)d_in[1];
  const float* mask = (const float*)d_in[2];
  const float* we   = (const float*)d_in[3];
  const float* se   = (const float*)d_in[4];
  const float* pe   = (const float*)d_in[5];
  const float* ln0g = (const float*)d_in[6];
  const float* ln0b = (const float*)d_in[7];
  const float* Wq   = (const float*)d_in[8];
  const float* Wk   = (const float*)d_in[9];
  const float* Wv   = (const float*)d_in[10];
  const float* W1   = (const float*)d_in[11];
  const float* b1   = (const float*)d_in[12];
  const float* Wi   = (const float*)d_in[13];
  const float* bi   = (const float*)d_in[14];
  const float* W2   = (const float*)d_in[15];
  const float* b2   = (const float*)d_in[16];
  const float* lng  = (const float*)d_in[17];
  const float* lnb  = (const float*)d_in[18];
  const float* Wp   = (const float*)d_in[19];
  const float* bp   = (const float*)d_in[20];

  const long long BSD = (long long)Bx * Sx * Dx;       // 3,145,728
  const long long BSI = (long long)Bx * Sx * Ix;       // 12,582,912
  const long long SCO = (long long)Bx * Hx * Sx * Sx;  // 25,165,824
  const long long WDD = (long long)Lx * Dx * Dx;
  const long long WDI = (long long)Lx * Dx * Ix;

  char* ws = (char*)d_ws;
  size_t off = 0;
  auto alloc = [&](long long bytes) -> char* {
    char* p = ws + off;
    off = (off + (size_t)bytes + 255) & ~(size_t)255;
    return p;
  };

  // bf16 transposed weights [L,N,K]
  __bf16* wqb = (__bf16*)alloc(WDD * 2);
  __bf16* wkb = (__bf16*)alloc(WDD * 2);
  __bf16* wvb = (__bf16*)alloc(WDD * 2);
  __bf16* w1b = (__bf16*)alloc(WDD * 2);
  __bf16* wib = (__bf16*)alloc(WDI * 2);
  __bf16* w2b = (__bf16*)alloc(WDI * 2);
  float*  xf  = (float*) alloc(BSD * 4);
  __bf16* xb  = (__bf16*)alloc(BSD * 2);
  __bf16* qt  = (__bf16*)alloc(BSD * 2);
  __bf16* kt  = (__bf16*)alloc(BSD * 2);
  __bf16* vt  = (__bf16*)alloc(BSD * 2);
  __bf16* qh  = (__bf16*)alloc(BSD * 2);
  __bf16* kh  = (__bf16*)alloc(BSD * 2);
  __bf16* vT  = (__bf16*)alloc(BSD * 2);
  float*  sc  = (float*) alloc(SCO * 4);
  __bf16* pr  = (__bf16*)alloc(SCO * 2);
  __bf16* ab  = (__bf16*)alloc(BSD * 2);
  __bf16* ctx = (__bf16*)alloc(BSD * 2);
  __bf16* h1b = (__bf16*)alloc(BSD * 2);
  __bf16* h2b = (__bf16*)alloc(BSI * 2);
  float*  h3f = (float*) alloc(BSD * 4);

  // ---- weights: fp32 [L,K,N] -> bf16 [L,N,K] ----
  const dim3 tDD(Dx / 32, Dx / 32, Lx);
  const dim3 tDI(Ix / 32, Dx / 32, Lx);   // Wi: K=D, N=I
  const dim3 tID(Dx / 32, Ix / 32, Lx);   // W2: K=I, N=D
  transpose_f2b_kernel<<<tDD, 256, 0, stream>>>(Wq, wqb, Dx, Dx);
  transpose_f2b_kernel<<<tDD, 256, 0, stream>>>(Wk, wkb, Dx, Dx);
  transpose_f2b_kernel<<<tDD, 256, 0, stream>>>(Wv, wvb, Dx, Dx);
  transpose_f2b_kernel<<<tDD, 256, 0, stream>>>(W1, w1b, Dx, Dx);
  transpose_f2b_kernel<<<tDI, 256, 0, stream>>>(Wi, wib, Dx, Ix);
  transpose_f2b_kernel<<<tID, 256, 0, stream>>>(W2, w2b, Ix, Dx);

  // ---- embedding + initial 2-D layernorm ----
  const int eg = (int)((BSD + 255) / 256);
  embed_kernel<<<eg, 256, 0, stream>>>(ids, seg, we, se, pe, h3f);
  ln2d_kernel<<<Bx, 512, 0, stream>>>(h3f, nullptr, ln0g, ln0b, xf, xb);

  const dim3 gDD(Dx / 128, (Bx * Sx) / 128, 1);   // N=768
  const dim3 gDI(Ix / 128, (Bx * Sx) / 128, 1);   // N=3072
  const dim3 gSc(Sx / 128, Sx / 128, Bx * Hx);    // scores 512x512 x96
  const dim3 gCt(DHx / 64, Sx / 128, Bx * Hx);    // context 512x64 x96

  for (int l = 0; l < Lx; ++l) {
    // Q, K, V projections (no bias)
    gemm_bt_kernel<128, 2, 4><<<gDD, 256, 0, stream>>>(
        xb, wqb + (long long)l * Dx * Dx, nullptr, nullptr, qt,
        Bx * Sx, Dx, Dx, 0, 0, 0, 0);
    gemm_bt_kernel<128, 2, 4><<<gDD, 256, 0, stream>>>(
        xb, wkb + (long long)l * Dx * Dx, nullptr, nullptr, kt,
        Bx * Sx, Dx, Dx, 0, 0, 0, 0);
    gemm_bt_kernel<128, 2, 4><<<gDD, 256, 0, stream>>>(
        xb, wvb + (long long)l * Dx * Dx, nullptr, nullptr, vt,
        Bx * Sx, Dx, Dx, 0, 0, 0, 0);
    qkv_repack_kernel<<<eg, 256, 0, stream>>>(qt, kt, vt, qh, kh, vT);

    // scores = q @ k^T  (BT operand is k itself, [S,DH] per head)
    gemm_bt_kernel<128, 2, 4><<<gSc, 256, 0, stream>>>(
        qh, kh, nullptr, sc, nullptr, Sx, Sx, DHx,
        (long long)Sx * DHx, (long long)Sx * DHx, (long long)Sx * Sx, 0);
    softmax_kernel<<<Bx * Hx * Sx, 256, 0, stream>>>(sc, mask, pr, 0.125f); // 1/sqrt(64)

    // context = probs @ v  (BT operand is v^T, [DH,S] per head)
    gemm_bt_kernel<64, 4, 2><<<gCt, 256, 0, stream>>>(
        pr, vT, nullptr, nullptr, ab, Sx, DHx, Sx,
        (long long)Sx * Sx, (long long)DHx * Sx, (long long)Sx * DHx, 0);
    attn_repack_kernel<<<eg, 256, 0, stream>>>(ab, ctx);

    // FFN chain: relu(ctx@W1+b1) -> relu(@Wi+bi) -> relu(@W2+b2)
    gemm_bt_kernel<128, 2, 4><<<gDD, 256, 0, stream>>>(
        ctx, w1b + (long long)l * Dx * Dx, b1 + l * Dx, nullptr, h1b,
        Bx * Sx, Dx, Dx, 0, 0, 0, 1);
    gemm_bt_kernel<128, 2, 4><<<gDI, 256, 0, stream>>>(
        h1b, wib + (long long)l * Dx * Ix, bi + l * Ix, nullptr, h2b,
        Bx * Sx, Ix, Dx, 0, 0, 0, 1);
    gemm_bt_kernel<128, 2, 4><<<gDD, 256, 0, stream>>>(
        h2b, w2b + (long long)l * Ix * Dx, b2 + l * Dx, h3f, nullptr,
        Bx * Sx, Dx, Ix, 0, 0, 0, 1);

    // x = ln2d(h + prev)
    ln2d_kernel<<<Bx, 512, 0, stream>>>(h3f, xf,
                                        lng + (long long)l * Sx * Dx,
                                        lnb + (long long)l * Sx * Dx, xf, xb);
  }

  // ---- pooled logits ----
  final_kernel<<<(2 * Bx * Sx + 255) / 256, 256, 0, stream>>>(xf, Wp, bp, mask, (float*)d_out);
}
